// MGTA_51135880626556
// MI455X (gfx1250) — matrix operand order repaired and checked by reading, not compiled, not run on hardware
//
#include <hip/hip_runtime.h>
#include <hip/hip_bf16.h>

typedef __attribute__((ext_vector_type(16))) _Float16 v16h;
typedef __attribute__((ext_vector_type(8)))  _Float16 v8h;
typedef __attribute__((ext_vector_type(8)))  float    v8f;

#define N_NODES 3072
#define NFEAT   128
#define NHID    64
#define NHEADS  8
#define LENGTH  8
#define NPATHS  2048
#define NROWS   16384      // NPATHS*LENGTH
#define NTASK   4
#define HF      512        // NHEADS*NHID
#define NPH     16         // 2 passes * 8 heads
#define ALPHA   0.2f

// ---- workspace layout (bytes) ----
#define OFF_FEAT16 ((size_t)0)                       // 3072*128*2        = 786432
#define OFF_WPACK  (OFF_FEAT16 + 786432)             // 16*4*4*32*16*2    = 262144
#define OFF_WH     (OFF_WPACK  + 262144)             // 16*3072*64*2      = 6291456
#define OFF_WHB    (OFF_WH     + 6291456)            // same size packed  = 6291456
#define OFF_F1     (OFF_WHB    + 6291456)            // 16*3072*4         = 196608
#define OFF_F2     (OFF_F1     + 196608)             // 196608
#define OFF_MASK   (OFF_F2     + 196608)             // 3072*96*4         = 1179648
#define OFF_X16    (OFF_MASK   + 1179648)            // 2*3072*512*2      = 6291456
#define OFF_GT     (OFF_X16    + 6291456)            // 2*512*16384*2     = 33554432
#define OFF_ACC    (OFF_GT     + 33554432)           // 2 floats

__device__ __forceinline__ v16h make16(v8h lo, v8h hi) {
    v16h r;
#pragma unroll
    for (int i = 0; i < 8; i++) { r[i] = lo[i]; r[i + 8] = hi[i]; }
    return r;
}

__device__ __forceinline__ v8f wmma_f16(v16h a, v16h b, v8f c) {
    return __builtin_amdgcn_wmma_f32_16x16x32_f16(false, a, false, b, (short)0, c, false, false);
}

// ---------------- setup kernels ----------------

__global__ void k_init_acc(float* acc) {
    if (threadIdx.x < 2) acc[threadIdx.x] = 0.f;
}

__global__ void k_feat_f16(const float* __restrict__ src, _Float16* __restrict__ dst, int n) {
    int t = blockIdx.x * blockDim.x + threadIdx.x;
    if (t < n) dst[t] = (_Float16)src[t];
}

// Pack W (f32 [8][128][64] x2) into B-fragment layout:
// wpack[((ph*4+nt)*4+kc)*32+lane][e] where e -> K = kc*32 + khalf*16 + e, col = nt*16 + lane%16
__global__ void k_wpack(const float* __restrict__ Wp, const float* __restrict__ Ws,
                        _Float16* __restrict__ wpack) {
    int t = blockIdx.x * blockDim.x + threadIdx.x;
    if (t >= NPH * 4 * 4 * 32) return;
    int lane = t & 31, kc = (t >> 5) & 3, nt = (t >> 7) & 3, ph = t >> 9;
    const float* W = (ph < 8) ? Wp : Ws;
    int h = ph & 7;
    int col = nt * 16 + (lane & 15);
    int khalf = lane >> 4;
    _Float16* dst = wpack + (size_t)t * 16;
    for (int e = 0; e < 16; e++) {
        int K = kc * 32 + khalf * 16 + e;
        dst[e] = (_Float16)W[(size_t)h * NFEAT * NHID + (size_t)K * NHID + col];
    }
}

// adjacency -> bitmask: mask[row][w] bit b = adj[row][w*32+b] > 0
__global__ void k_mask(const float* __restrict__ adj, unsigned* __restrict__ mask) {
    int t = blockIdx.x * blockDim.x + threadIdx.x;
    if (t >= N_NODES * 96) return;
    int row = t / 96, w = t % 96;
    unsigned bits = 0;
    const float* base = adj + (size_t)row * N_NODES + w * 32;
    for (int b = 0; b < 32; b++) bits |= (base[b] > 0.f) ? (1u << b) : 0u;
    mask[t] = bits;
}

// ---------------- stage 1: Wh = feat @ W, plus f1/f2 ----------------
// grid (192, 16), block 32. One wave = 16 rows x 64 cols, K=128.
__global__ __launch_bounds__(32) void k_gemm_wh(
    const _Float16* __restrict__ feat16, const _Float16* __restrict__ wpack,
    const float* __restrict__ ap, const float* __restrict__ as,
    _Float16* __restrict__ Wh, float* __restrict__ f1, float* __restrict__ f2) {
    int rb = blockIdx.x, ph = blockIdx.y, lane = threadIdx.x;
    int m = lane & 15, khalf = lane >> 4;

    // A fragments: features rows rb*16..+15, K-chunks of 32
    v16h A[4];
    const _Float16* frow = feat16 + (size_t)(rb * 16 + m) * NFEAT;
#pragma unroll
    for (int kc = 0; kc < 4; kc++) {
        v8h lo = *(const v8h*)(frow + kc * 32 + khalf * 16);
        v8h hi = *(const v8h*)(frow + kc * 32 + khalf * 16 + 8);
        A[kc] = make16(lo, hi);
    }

    v8f acc[4];
#pragma unroll
    for (int nt = 0; nt < 4; nt++) {
        v8f c = {};
#pragma unroll
        for (int kc = 0; kc < 4; kc++) {
            v16h b = *(const v16h*)(wpack + ((size_t)((ph * 4 + nt) * 4 + kc) * 32 + lane) * 16);
            c = wmma_f16(A[kc], b, c);
        }
        acc[nt] = c;
    }

    // store Wh (f16): D layout -> lane holds col (nt*16+m), rows khalf*8 + r
#pragma unroll
    for (int nt = 0; nt < 4; nt++) {
#pragma unroll
        for (int r = 0; r < 8; r++) {
            int node = rb * 16 + khalf * 8 + r;
            Wh[(size_t)ph * N_NODES * NHID + (size_t)node * NHID + nt * 16 + m] =
                (_Float16)acc[nt][r];
        }
    }

    // f1/f2: per-row dot with attention vectors a[:64], a[64:]
    const float* avec = ((ph < 8) ? ap : as) + (size_t)(ph & 7) * 128;
    float p1[8], p2[8];
#pragma unroll
    for (int r = 0; r < 8; r++) { p1[r] = 0.f; p2[r] = 0.f; }
#pragma unroll
    for (int nt = 0; nt < 4; nt++) {
        float a1 = avec[nt * 16 + m];
        float a2 = avec[64 + nt * 16 + m];
#pragma unroll
        for (int r = 0; r < 8; r++) { p1[r] += acc[nt][r] * a1; p2[r] += acc[nt][r] * a2; }
    }
#pragma unroll
    for (int off = 8; off >= 1; off >>= 1) {
#pragma unroll
        for (int r = 0; r < 8; r++) {
            p1[r] += __shfl_xor(p1[r], off, 32);
            p2[r] += __shfl_xor(p2[r], off, 32);
        }
    }
    if (m == 0) {
#pragma unroll
        for (int r = 0; r < 8; r++) {
            int node = rb * 16 + khalf * 8 + r;
            f1[(size_t)ph * N_NODES + node] = p1[r];
            f2[(size_t)ph * N_NODES + node] = p2[r];
        }
    }
}

// repack Wh into B-fragment layout: whb[((ph*96+jb)*4+nt)*32+lane][e], K=node=jb*32+khalf*8+e (e<8), jb*32+16+khalf*8+(e-8) (e>=8)
__global__ void k_whb(const _Float16* __restrict__ Wh, _Float16* __restrict__ whb) {
    int t = blockIdx.x * blockDim.x + threadIdx.x;
    if (t >= NPH * 96 * 4 * 32) return;
    int lane = t & 31, nt = (t >> 5) & 3;
    int rest = t >> 7;
    int jb = rest % 96, ph = rest / 96;
    int col = nt * 16 + (lane & 15), khalf = lane >> 4;
    _Float16* dst = whb + (size_t)t * 16;
    for (int e = 0; e < 16; e++) {
        int node = jb * 32 + ((e < 8) ? (khalf * 8 + e) : (16 + khalf * 8 + (e - 8)));
        dst[e] = Wh[(size_t)ph * N_NODES * NHID + (size_t)node * NHID + col];
    }
}

// ---------------- stage 2: fused masked softmax attention + att@Wh + elu ----------------
// grid (192, 16), block 32. One wave = 16 output rows, streams 96 column tiles of 32.
__global__ __launch_bounds__(32) void k_attn(
    const _Float16* __restrict__ whb, const float* __restrict__ f1,
    const float* __restrict__ f2, const unsigned* __restrict__ mask,
    _Float16* __restrict__ xf16) {
    int rb = blockIdx.x, ph = blockIdx.y, lane = threadIdx.x;
    int m = lane & 15, khalf = lane >> 4, base = khalf * 8;

    float f1v = f1[(size_t)ph * N_NODES + rb * 16 + m];
    const unsigned* mrow = mask + (size_t)(rb * 16 + m) * 96;
    const float* f2p = f2 + (size_t)ph * N_NODES;
    const _Float16* whbp = whb + (size_t)ph * 96 * 4 * 32 * 16;

    v8f acc[4];
#pragma unroll
    for (int nt = 0; nt < 4; nt++) acc[nt] = (v8f){};
    float den = 0.f;

    for (int jb = 0; jb < 96; jb++) {
        unsigned mw = mrow[jb];
        v8f r0 = *(const v8f*)(f2p + jb * 32 + base);
        v8f r1 = *(const v8f*)(f2p + jb * 32 + base + 16);
        v16h aw;
#pragma unroll
        for (int e = 0; e < 16; e++) {
            int K = (e < 8) ? (base + e) : (base + 16 + (e - 8));
            float fe = (e < 8) ? r0[e] : r1[e - 8];
            float s = f1v + fe;
            float el = (s > 0.f) ? s : (ALPHA * s);
            float w = ((mw >> K) & 1u) ? __expf(el) : 0.f;
            den += w;
            aw[e] = (_Float16)w;
        }
        const _Float16* bb = whbp + (size_t)jb * 4 * 32 * 16;
#pragma unroll
        for (int nt = 0; nt < 4; nt++) {
            v16h b = *(const v16h*)(bb + ((size_t)nt * 32 + lane) * 16);
            acc[nt] = wmma_f16(aw, b, acc[nt]);
        }
    }

    float dtot = den + __shfl_xor(den, 16, 32);  // lanes m and m+16 cover disjoint K subsets
    int pass = ph >> 3, h = ph & 7;
#pragma unroll
    for (int r = 0; r < 8; r++) {
        int rowm = khalf * 8 + r;
        float dr = __shfl(dtot, rowm, 32);  // lane 'rowm' (<16) holds that row's denominator
        int node = rb * 16 + rowm;
#pragma unroll
        for (int nt = 0; nt < 4; nt++) {
            float v = acc[nt][r] / dr;
            float o = (v > 0.f) ? v : (__expf(v) - 1.f);  // elu
            xf16[(size_t)pass * N_NODES * HF + (size_t)node * HF + h * NHID + nt * 16 + m] =
                (_Float16)o;
        }
    }
}

// ---------------- stage 3: transpose-gather gT[p][c][r] = x_f16[p][path[r]][c] ----------------
__global__ void k_gatherT(const _Float16* __restrict__ xf16, const int* __restrict__ path,
                          _Float16* __restrict__ gT) {
    size_t t = (size_t)blockIdx.x * blockDim.x + threadIdx.x;  // 2*512*16384 threads
    int r = (int)(t & 16383);
    int c = (int)((t >> 14) & 511);
    int p = (int)(t >> 23);
    if (p >= 2) return;
    int node = path[r];
    gT[t] = xf16[(size_t)p * N_NODES * HF + (size_t)node * HF + c];
}

// ---------------- stage 4: adversarial loss ----------------
__global__ __launch_bounds__(256) void k_adv(
    const _Float16* __restrict__ xf16, const int* __restrict__ path,
    const int* __restrict__ task, const float* __restrict__ Wsc,
    const float* __restrict__ bsc, float* __restrict__ acc) {
    __shared__ float sd[256];
    int r = blockIdx.x * 256 + threadIdx.x;
    float contrib = 0.f;
    if (r < NROWS) {
        int node = path[r];
        const _Float16* row = xf16 + (size_t)1 * N_NODES * HF + (size_t)node * HF;
        float lg[4] = {bsc[0], bsc[1], bsc[2], bsc[3]};
        for (int c8 = 0; c8 < HF / 8; c8++) {
            v8h v = *(const v8h*)(row + c8 * 8);
#pragma unroll
            for (int e = 0; e < 8; e++) {
                float sv = (float)v[e];
                const float* wr = Wsc + (size_t)(c8 * 8 + e) * NTASK;
                lg[0] += sv * wr[0]; lg[1] += sv * wr[1];
                lg[2] += sv * wr[2]; lg[3] += sv * wr[3];
            }
        }
        float z[4], mx = -1e30f;
#pragma unroll
        for (int t2 = 0; t2 < 4; t2++) {
            z[t2] = 1.f / (1.f + __expf(-lg[t2]));
            mx = fmaxf(mx, z[t2]);
        }
        float ex[4], S = 0.f;
#pragma unroll
        for (int t2 = 0; t2 < 4; t2++) { ex[t2] = __expf(z[t2] - mx); S += ex[t2]; }
        float pr[4], mx2 = -1e30f;
#pragma unroll
        for (int t2 = 0; t2 < 4; t2++) { pr[t2] = ex[t2] / S; mx2 = fmaxf(mx2, pr[t2]); }
        float S2 = 0.f;
#pragma unroll
        for (int t2 = 0; t2 < 4; t2++) S2 += __expf(pr[t2] - mx2);
        float lse = mx2 + __logf(S2);
        int tk = task[r];
        contrib = -(pr[tk] - lse);  // -log_softmax(node_task)[task]
    }
    sd[threadIdx.x] = contrib;
    __syncthreads();
    for (int s = 128; s > 0; s >>= 1) {
        if (threadIdx.x < s) sd[threadIdx.x] += sd[threadIdx.x + s];
        __syncthreads();
    }
    if (threadIdx.x == 0) atomicAdd(acc, sd[0]);
}

// ---------------- stage 5: diff loss  ||S^T P||_F^2  via WMMA ----------------
// grid (32, 32), block 32. D tile [16 share-cols x 16 private-cols], K = 16384 gathered rows.
__global__ __launch_bounds__(32) void k_diff(const _Float16* __restrict__ gT,
                                             float* __restrict__ acc) {
    int tm = blockIdx.x, tn = blockIdx.y, lane = threadIdx.x;
    int m = lane & 15, khalf = lane >> 4, base = khalf * 8;
    const _Float16* gS = gT + (size_t)1 * HF * NROWS;  // share  (transposed: [512][16384])
    const _Float16* gP = gT;                           // private
    const _Float16* arow = gS + (size_t)(tm * 16 + m) * NROWS;
    const _Float16* brow = gP + (size_t)(tn * 16 + m) * NROWS + khalf * 16;
    v8f c = {};
    for (int kb = 0; kb < NROWS / 32; kb++) {
        v8h lo = *(const v8h*)(arow + kb * 32 + khalf * 16);
        v8h hi = *(const v8h*)(arow + kb * 32 + khalf * 16 + 8);
        v16h a = make16(lo, hi);
        v16h b = *(const v16h*)(brow + kb * 32);
        c = wmma_f16(a, b, c);
    }
    float s = 0.f;
#pragma unroll
    for (int r = 0; r < 8; r++) s += c[r] * c[r];
#pragma unroll
    for (int off = 16; off >= 1; off >>= 1) s += __shfl_xor(s, off, 32);
    if (lane == 0) atomicAdd(acc, s);
}

// ---------------- stage 6: final classifier x = sigmoid(concat @ Wc + bc) ----------------
// one wave per path (8 waves per 256-thread block)
__global__ __launch_bounds__(256) void k_final(
    const _Float16* __restrict__ xf16, const int* __restrict__ path,
    const float* __restrict__ Wc, const float* __restrict__ bc,
    float* __restrict__ out) {
    int wave = threadIdx.x >> 5, lane = threadIdx.x & 31;
    int p = blockIdx.x * 8 + wave;
    float sum = 0.f;
    int c0 = lane * 16;
    for (int l = 0; l < LENGTH; l++) {
        int node = path[p * LENGTH + l];
        const _Float16* sh = xf16 + (size_t)1 * N_NODES * HF + (size_t)node * HF + c0;
        const _Float16* pv = xf16 + (size_t)node * HF + c0;
        const float* wS = Wc + (size_t)l * 2 * HF + c0;
        const float* wP = Wc + (size_t)l * 2 * HF + HF + c0;
#pragma unroll
        for (int e = 0; e < 16; e++) {
            sum += (float)sh[e] * wS[e];
            sum += (float)pv[e] * wP[e];
        }
    }
#pragma unroll
    for (int off = 16; off >= 1; off >>= 1) sum += __shfl_xor(sum, off, 32);
    if (lane == 0) out[p] = 1.f / (1.f + __expf(-(sum + bc[0])));
}

__global__ void k_finalize(const float* __restrict__ acc, float* __restrict__ out) {
    if (threadIdx.x == 0) {
        out[NPATHS] = acc[0] / (float)NROWS;  // adv_loss (mean)
        out[NPATHS + 1] = acc[1];             // diff_loss
    }
}

// ---------------- host launch ----------------
extern "C" void kernel_launch(void* const* d_in, const int* in_sizes, int n_in,
                              void* d_out, int out_size, void* d_ws, size_t ws_size,
                              hipStream_t stream) {
    const float* features = (const float*)d_in[0];
    const float* adj      = (const float*)d_in[1];
    const int*   path     = (const int*)d_in[2];
    const int*   task     = (const int*)d_in[3];
    const float* Wp       = (const float*)d_in[4];
    const float* ap       = (const float*)d_in[5];
    const float* Ws       = (const float*)d_in[6];
    const float* a_s      = (const float*)d_in[7];
    const float* Wsc      = (const float*)d_in[8];
    const float* bsc      = (const float*)d_in[9];
    const float* Wc       = (const float*)d_in[10];
    const float* bc       = (const float*)d_in[11];
    float* out = (float*)d_out;

    char* ws = (char*)d_ws;
    _Float16* feat16 = (_Float16*)(ws + OFF_FEAT16);
    _Float16* wpack  = (_Float16*)(ws + OFF_WPACK);
    _Float16* Wh     = (_Float16*)(ws + OFF_WH);
    _Float16* whb    = (_Float16*)(ws + OFF_WHB);
    float*    f1     = (float*)(ws + OFF_F1);
    float*    f2     = (float*)(ws + OFF_F2);
    unsigned* mask   = (unsigned*)(ws + OFF_MASK);
    _Float16* xf16   = (_Float16*)(ws + OFF_X16);
    _Float16* gT     = (_Float16*)(ws + OFF_GT);
    float*    accb   = (float*)(ws + OFF_ACC);

    k_init_acc<<<1, 32, 0, stream>>>(accb);
    k_feat_f16<<<(N_NODES * NFEAT + 255) / 256, 256, 0, stream>>>(features, feat16,
                                                                  N_NODES * NFEAT);
    k_wpack<<<(NPH * 4 * 4 * 32 + 255) / 256, 256, 0, stream>>>(Wp, Ws, wpack);
    k_mask<<<(N_NODES * 96 + 255) / 256, 256, 0, stream>>>(adj, mask);

    k_gemm_wh<<<dim3(N_NODES / 16, NPH), 32, 0, stream>>>(feat16, wpack, ap, a_s, Wh, f1, f2);
    k_whb<<<(NPH * 96 * 4 * 32 + 255) / 256, 256, 0, stream>>>(Wh, whb);

    k_attn<<<dim3(N_NODES / 16, NPH), 32, 0, stream>>>(whb, f1, f2, mask, xf16);

    k_gatherT<<<(2u * HF * NROWS) / 256, 256, 0, stream>>>(xf16, path, gT);

    k_adv<<<NROWS / 256, 256, 0, stream>>>(xf16, path, task, Wsc, bsc, accb);
    k_diff<<<dim3(HF / 16, HF / 16), 32, 0, stream>>>(gT, accb + 1);
    k_final<<<NPATHS / 8, 256, 0, stream>>>(xf16, path, Wc, bc, out);
    k_finalize<<<1, 32, 0, stream>>>(accb, out);
}
